// VectorQuantizer_49177375539633
// MI455X (gfx1250) — compile-verified, matrix operand order checked by reading, not run on hardware
//
#include <hip/hip_runtime.h>
#include <math.h>

// ---------------------------------------------------------------------------
// Multi-scale residual soft VQ (VAR-style), MI455X / gfx1250, wave32 + WMMA.
//
// Shapes: B=32, N=1024, C=512, K=4096, 11 scales pn=1..1024.
// Per scale: logits = w_sq - 2 * z @ W^T  (M x K, inner C)
//            probs  = softmax(logits) over K
//            h      = probs @ W           (M x C, inner K)
// Fused per 32-row tile in one workgroup using v_wmma_f32_16x16x32_f16.
// The 32 x 4096 f16 logits tile (256 KB) + 32 x 512 f16 z tile (32 KB) live
// in LDS (288 KB of the 320 KB/WGP CDNA5 budget) -> each B-fragment load is
// reused by two WMMAs, halving L2 traffic vs a 16-row tile.
// ---------------------------------------------------------------------------

typedef __attribute__((ext_vector_type(16))) _Float16 v16h;
typedef __attribute__((ext_vector_type(8)))  _Float16 v8h;
typedef __attribute__((ext_vector_type(8)))  float    v8f;

#define B_   32
#define N_   1024
#define C_   512
#define K_   4096
#define SN_  11
#define BNC_ (B_ * N_ * C_)

union V16U { v16h v; v8h h[2]; };

// -------------------------------------------------------------------- init
__global__ void k_init(const float* __restrict__ f,
                       float* __restrict__ f_rest,
                       float* __restrict__ f_hat,
                       double* __restrict__ acc) {
  int i = blockIdx.x * blockDim.x + threadIdx.x;
  if (i == 0) *acc = 0.0;
  if (i < BNC_) { f_rest[i] = f[i]; f_hat[i] = 0.0f; }
}

// --------------------------------------------------- codebook prep: wsq, f16
__global__ void k_prepw(const float* __restrict__ W,
                        _Float16* __restrict__ Wh,   // (K, C) f16
                        _Float16* __restrict__ Wt,   // (C, K) f16
                        float* __restrict__ wsq) {
  __shared__ float red[256];
  const int k = blockIdx.x;
  float s = 0.0f;
  for (int c = threadIdx.x; c < C_; c += 256) {
    float w = W[(size_t)k * C_ + c];
    Wh[(size_t)k * C_ + c] = (_Float16)w;
    Wt[(size_t)c * K_ + k] = (_Float16)w;
    s += w * w;
  }
  red[threadIdx.x] = s;
  __syncthreads();
  for (int off = 128; off > 0; off >>= 1) {
    if (threadIdx.x < off) red[threadIdx.x] += red[threadIdx.x + off];
    __syncthreads();
  }
  if (threadIdx.x == 0) wsq[k] = red[0];
}

// ------------------------- downsample, direct path (pn >= 32 -> sN <= 32) ---
__global__ void k_down_direct(const float* __restrict__ f_rest,
                              _Float16* __restrict__ z, int pn) {
  int i = blockIdx.x * blockDim.x + threadIdx.x;  // over (B*pn)*C
  int M = B_ * pn;
  if (i >= M * C_) return;
  int c = i % C_;
  int m = i / C_;
  int b = m / pn, p = m % pn;
  int sN = N_ / pn;
  const float* src = f_rest + ((size_t)(b * N_ + p * sN)) * C_ + c;
  float s = 0.0f;
  for (int j = 0; j < sN; ++j) s += src[(size_t)j * C_];
  z[i] = (_Float16)(s / (float)sN);
}

// ------------- downsample stage 1 for pn < 32: block-mean-of-32 over N ------
// reads the full 64MB residual with 524288 coalesced threads (BW-bound pass)
__global__ void k_mean32(const float* __restrict__ f_rest,
                         float* __restrict__ g32) {   // (B, 32, C)
  int i = blockIdx.x * blockDim.x + threadIdx.x;      // B*32*C
  int c = i % C_;
  int m = i / C_;              // b*32 + p
  int b = m / 32, p = m % 32;
  const float* src = f_rest + ((size_t)(b * N_ + p * 32)) * C_ + c;
  float s = 0.0f;
  for (int j = 0; j < 32; ++j) s += src[(size_t)j * C_];
  g32[i] = s * (1.0f / 32.0f);
}

// ------------- downsample stage 2 for pn < 32: mean of 32/pn g32 rows -------
__global__ void k_down_from32(const float* __restrict__ g32,
                              _Float16* __restrict__ z, int pn) {
  int i = blockIdx.x * blockDim.x + threadIdx.x;
  int M = B_ * pn;
  if (i >= M * C_) return;
  int c = i % C_;
  int m = i / C_;
  int b = m / pn, p = m % pn;
  int r = 32 / pn;
  const float* src = g32 + ((size_t)(b * 32 + p * r)) * C_ + c;
  float s = 0.0f;
  for (int j = 0; j < r; ++j) s += src[(size_t)j * C_];
  z[i] = (_Float16)(s / (float)r);
}

// ---------------- fused: GEMM1 (z@W^T) -> softmax over K -> GEMM2 (p@W) -----
// grid.x = M/32 tiles of 32 rows; 256 threads = 8 wave32.
__global__ __launch_bounds__(256) void k_quant(
    const _Float16* __restrict__ z,   // (M, C) f16
    const _Float16* __restrict__ Wh,  // (K, C) f16
    const _Float16* __restrict__ Wt,  // (C, K) f16
    const float* __restrict__ wsq,    // (K,)
    float* __restrict__ h) {          // (M, C) f32 out
  extern __shared__ char smem[];
  _Float16* zA = (_Float16*)smem;                    // 32 x C   (32 KB)
  _Float16* pL = (_Float16*)(smem + 32 * C_ * 2);    // 32 x K   (256 KB)

  const int tid  = threadIdx.x;
  const int lane = tid & 31;
  const int wave = tid >> 5;
  const int rowBase = blockIdx.x * 32;

  // stage the 32 x C z-tile into LDS (32 KB) with 128-bit copies
  {
    const uint4* src = (const uint4*)(z + (size_t)rowBase * C_);
    uint4* dst = (uint4*)zA;
    for (int i = tid; i < 32 * C_ * 2 / 16; i += 256) dst[i] = src[i];
  }
  __syncthreads();

  const int hi   = lane >> 4;   // K-half select per WMMA fragment layout
  const int arow = lane & 15;   // A-fragment row (M)
  const int col  = lane & 15;   // B-fragment column (N)

  // ---- Phase 1: logits = wsq - 2 * z @ W^T ; wave owns K-slab [wave*512,+512)
  //      two M-tiles share each B fragment (2 WMMA per W load)
  for (int kt = 0; kt < 32; ++kt) {
    const int k0 = wave * 512 + kt * 16;
    v8f acc0 = {}, acc1 = {};
    for (int c0 = 0; c0 < C_; c0 += 32) {
      V16U a0, a1, b;
      const _Float16* wr = Wh + (size_t)(k0 + col) * C_ + c0;
      b.h[0] = *(const v8h*)(wr + hi * 8);
      b.h[1] = *(const v8h*)(wr + 16 + hi * 8);
      const _Float16* zr0 = zA + (size_t)arow * C_ + c0;
      a0.h[0] = *(const v8h*)(zr0 + hi * 8);
      a0.h[1] = *(const v8h*)(zr0 + 16 + hi * 8);
      const _Float16* zr1 = zr0 + 16 * C_;
      a1.h[0] = *(const v8h*)(zr1 + hi * 8);
      a1.h[1] = *(const v8h*)(zr1 + 16 + hi * 8);
      acc0 = __builtin_amdgcn_wmma_f32_16x16x32_f16(
          false, a0.v, false, b.v, (short)0, acc0, false, false);
      acc1 = __builtin_amdgcn_wmma_f32_16x16x32_f16(
          false, a1.v, false, b.v, (short)0, acc1, false, false);
    }
    float s = wsq[k0 + col];
#pragma unroll
    for (int j = 0; j < 8; ++j) {
      pL[(j + 8 * hi) * K_ + (k0 + col)]        = (_Float16)(s - 2.0f * acc0[j]);
      pL[(16 + j + 8 * hi) * K_ + (k0 + col)]   = (_Float16)(s - 2.0f * acc1[j]);
    }
  }
  __syncthreads();

  // ---- softmax over K; wave w owns rows 4w .. 4w+3
  for (int r = wave * 4; r < wave * 4 + 4; ++r) {
    _Float16* row = pL + (size_t)r * K_;
    float mx = -1e30f;
    for (int k = lane; k < K_; k += 32) mx = fmaxf(mx, (float)row[k]);
    for (int m = 16; m > 0; m >>= 1) mx = fmaxf(mx, __shfl_xor(mx, m, 32));
    float sum = 0.0f;
    for (int k = lane; k < K_; k += 32) {
      float e = __expf((float)row[k] - mx);
      row[k] = (_Float16)e;
      sum += e;
    }
    for (int m = 16; m > 0; m >>= 1) sum += __shfl_xor(sum, m, 32);
    float inv = 1.0f / sum;
    for (int k = lane; k < K_; k += 32) row[k] = (_Float16)((float)row[k] * inv);
  }
  __syncthreads();

  // ---- Phase 2: h = p @ W ; wave owns C-slab [wave*64,+64), B reused 2x
  for (int nt = 0; nt < 4; ++nt) {
    const int c0n = wave * 64 + nt * 16;
    v8f acc0 = {}, acc1 = {};
    for (int kk = 0; kk < K_; kk += 32) {
      V16U a0, a1, b;
      const _Float16* wt = Wt + (size_t)(c0n + col) * K_ + kk;
      b.h[0] = *(const v8h*)(wt + hi * 8);
      b.h[1] = *(const v8h*)(wt + 16 + hi * 8);
      const _Float16* pr0 = pL + (size_t)arow * K_ + kk;
      a0.h[0] = *(const v8h*)(pr0 + hi * 8);
      a0.h[1] = *(const v8h*)(pr0 + 16 + hi * 8);
      const _Float16* pr1 = pr0 + 16 * K_;
      a1.h[0] = *(const v8h*)(pr1 + hi * 8);
      a1.h[1] = *(const v8h*)(pr1 + 16 + hi * 8);
      acc0 = __builtin_amdgcn_wmma_f32_16x16x32_f16(
          false, a0.v, false, b.v, (short)0, acc0, false, false);
      acc1 = __builtin_amdgcn_wmma_f32_16x16x32_f16(
          false, a1.v, false, b.v, (short)0, acc1, false, false);
    }
#pragma unroll
    for (int j = 0; j < 8; ++j) {
      h[(size_t)(rowBase + j + 8 * hi) * C_ + c0n + col]      = acc0[j];
      h[(size_t)(rowBase + 16 + j + 8 * hi) * C_ + c0n + col] = acc1[j];
    }
  }
}

// ---------------- linear upsample + residual update + squared-error reduce --
__global__ void k_up(const float* __restrict__ f,
                     const float* __restrict__ hbuf,   // (B*pn, C)
                     float* __restrict__ f_hat,
                     float* __restrict__ f_rest,
                     double* __restrict__ acc, int pn) {
  __shared__ float red[256];
  int i = blockIdx.x * blockDim.x + threadIdx.x;  // exactly BNC_
  int c = i % C_;
  int n = (i / C_) % N_;
  int b = i / (C_ * N_);
  float scale = (float)pn / (float)N_;
  float coord = ((float)n + 0.5f) * scale - 0.5f;
  coord = fminf(fmaxf(coord, 0.0f), (float)(pn - 1));
  int i0 = (int)coord;               // floor (coord >= 0)
  int i1 = min(i0 + 1, pn - 1);
  float w = coord - (float)i0;
  float hv = hbuf[(size_t)(b * pn + i0) * C_ + c] * (1.0f - w) +
             hbuf[(size_t)(b * pn + i1) * C_ + c] * w;
  float fh = f_hat[i] + hv;
  f_hat[i] = fh;
  f_rest[i] -= hv;
  float d = fh - f[i];
  red[threadIdx.x] = d * d;
  __syncthreads();
  for (int off = 128; off > 0; off >>= 1) {
    if (threadIdx.x < off) red[threadIdx.x] += red[threadIdx.x + off];
    __syncthreads();
  }
  if (threadIdx.x == 0) unsafeAtomicAdd(acc, (double)red[0]);
}

// ------------------------------------------------------------------- finish
__global__ void k_finish(const double* __restrict__ acc,
                         float* __restrict__ out_sc) {
  double q = *acc / ((double)BNC_ * (double)SN_);
  out_sc[0] = (float)(0.25 * q);  // commit
  out_sc[1] = (float)q;           // qlat
}

// ---------------------------------------------------------------------------
extern "C" void kernel_launch(void* const* d_in, const int* in_sizes, int n_in,
                              void* d_out, int out_size, void* d_ws, size_t ws_size,
                              hipStream_t stream) {
  const float* f = (const float*)d_in[0];   // (B, N, C)
  const float* W = (const float*)d_in[1];   // (K, C)
  float* f_hat  = (float*)d_out;            // first B*N*C elements
  float* out_sc = f_hat + BNC_;             // commit, qlat

  char* ws = (char*)d_ws;
  size_t off = 0;
  float*    f_rest = (float*)(ws + off);    off += (size_t)BNC_ * 4;
  float*    hbuf   = (float*)(ws + off);    off += (size_t)BNC_ * 4;       // max M*C
  _Float16* zb     = (_Float16*)(ws + off); off += (size_t)BNC_ * 2;       // max M*C
  float*    g32    = (float*)(ws + off);    off += (size_t)B_ * 32 * C_ * 4;
  _Float16* Wh     = (_Float16*)(ws + off); off += (size_t)K_ * C_ * 2;
  _Float16* Wt     = (_Float16*)(ws + off); off += (size_t)C_ * K_ * 2;
  float*    wsq    = (float*)(ws + off);    off += (size_t)K_ * 4;
  double*   acc    = (double*)(ws + off);   off += 8;

  k_init<<<BNC_ / 256, 256, 0, stream>>>(f, f_rest, f_hat, acc);
  k_prepw<<<K_, 256, 0, stream>>>(W, Wh, Wt, wsq);

  const size_t lds_bytes = (size_t)(32 * C_ + 32 * K_) * 2;  // 288 KB
  for (int s = 0; s < SN_; ++s) {
    int pn = 1 << s;
    int M = B_ * pn;
    if (pn >= 32) {
      k_down_direct<<<(M * C_ + 255) / 256, 256, 0, stream>>>(f_rest, zb, pn);
    } else {
      k_mean32<<<B_ * 32 * C_ / 256, 256, 0, stream>>>(f_rest, g32);
      k_down_from32<<<(M * C_ + 255) / 256, 256, 0, stream>>>(g32, zb, pn);
    }
    k_quant<<<M / 32, 256, lds_bytes, stream>>>(zb, Wh, Wt, wsq, hbuf);
    k_up<<<BNC_ / 256, 256, 0, stream>>>(f, hbuf, f_hat, f_rest, acc, pn);
  }
  k_finish<<<1, 1, 0, stream>>>(acc, out_sc);
}
